// DIFFN_52037823758410
// MI455X (gfx1250) — compile-verified, matrix operand order checked by reading, not run on hardware
//
#include <hip/hip_runtime.h>
#include <math.h>

typedef __attribute__((ext_vector_type(16))) _Float16 v16h;
typedef __attribute__((ext_vector_type(8)))  float    v8f;

#define DD 100
#define BB 128
#define NT 625          // 10000 / 16 N-tiles for hessian
#define NSLICE 5
#define NT_SLICE 125    // 625 / 5
#define KSTEPS 4        // K padded 100 -> 128 = 4 x 32
#define MT 7            // M tiles (100 -> 112)
#define FRAG_HALVES 512 // 32 lanes * 16 halves = 1 KB per fragment

// per-lane element -> K mapping inside a 32-wide K chunk (16-bit WMMA A/B layout):
// lane<16: e0..7 -> K 0..7,  e8..15 -> K 16..23
// lane>=16: e0..7 -> K 8..15, e8..15 -> K 24..31
__device__ __forceinline__ int klocal(int lane, int e) {
    return ((lane >> 4) << 3) + (e & 7) + ((e >> 3) << 4);
}

__device__ __forceinline__ v8f wmma_f16(v16h a, v16h b, v8f c) {
    return __builtin_amdgcn_wmma_f32_16x16x32_f16(false, a, false, b,
                                                  (short)0, c, false, false);
}

// ---------------------------------------------------------------------------
// Prep: fragment-ready f16 B matrices in workspace.
//   ws[0 .. NT*KSTEPS*512)              : P[k, n] = W1[k, n/100] * W1[k, n%100]
//   ws[NT*KSTEPS*512 .. +MT*KSTEPS*512) : W1[k, j]  (Jacobian GEMM)
// Zero-padded for k >= 100 (and j >= 100 for the W1 copy).
// ---------------------------------------------------------------------------
__global__ __launch_bounds__(256) void diffn_prep(const float* __restrict__ W1,
                                                  _Float16* __restrict__ ws) {
    int t = blockIdx.x * 256 + threadIdx.x;
    const int BH = NT * KSTEPS * 32;
    if (t < BH) {
        int f = t >> 5, lane = t & 31;
        int n_tile = f >> 2, kstep = f & 3;
        int n = n_tile * 16 + (lane & 15);
        int j = n / DD, l = n % DD;
        _Float16* dst = ws + (size_t)f * FRAG_HALVES + lane * 16;
        for (int e = 0; e < 16; ++e) {
            int k = kstep * 32 + klocal(lane, e);
            float v = (k < DD) ? W1[k * DD + j] * W1[k * DD + l] : 0.f;
            dst[e] = (_Float16)v;
        }
    } else {
        int u = t - BH;
        if (u < MT * KSTEPS * 32) {
            int f = u >> 5, lane = u & 31;
            int n_tile = f >> 2, kstep = f & 3;
            int j = n_tile * 16 + (lane & 15);
            _Float16* dst = ws + (size_t)(NT * KSTEPS) * FRAG_HALVES
                               + (size_t)f * FRAG_HALVES + lane * 16;
            for (int e = 0; e < 16; ++e) {
                int k = kstep * 32 + klocal(lane, e);
                float v = (k < DD && j < DD) ? W1[k * DD + j] : 0.f;
                dst[e] = (_Float16)v;
            }
        }
    }
}

// ---------------------------------------------------------------------------
// Main: one block per (sample, N-slice). 256 threads = 8 wave32s.
// ---------------------------------------------------------------------------
__global__ __launch_bounds__(256) void diffn_ffn(const float* __restrict__ x,
                                                 const float* __restrict__ W1,
                                                 const float* __restrict__ b1,
                                                 const float* __restrict__ W2,
                                                 const float* __restrict__ b2,
                                                 const _Float16* __restrict__ wsB,
                                                 float* __restrict__ out) {
    const int b = blockIdx.x;
    const int slice = blockIdx.y;
    const int tid = threadIdx.x;

    __shared__ float xs[DD], hs[DD], d1s[DD], d2s[DD];
    __shared__ __align__(32) _Float16 A1[MT * KSTEPS * FRAG_HALVES]; // W2*g'  (jac)
    __shared__ __align__(32) _Float16 A2[MT * KSTEPS * FRAG_HALVES]; // W2*g'' (hess)

    if (tid < DD) xs[tid] = x[b * DD + tid];
    __syncthreads();

    if (tid < DD) {
        float acc = b1[tid];
        const float* w = W1 + tid * DD;
        for (int j = 0; j < DD; ++j) acc += w[j] * xs[j];
        float z   = acc;
        float phi = 0.3989422804014327f * expf(-0.5f * z * z);
        float Phi = 0.5f * (1.f + erff(z * 0.7071067811865475f));
        hs[tid]  = z * Phi;               // gelu(z)
        d1s[tid] = Phi + z * phi;         // gelu'(z)
        d2s[tid] = (2.f - z * z) * phi;   // gelu''(z)
    }
    __syncthreads();

    if (slice == 0 && tid < DD) {
        float acc = b2[tid];
        const float* w = W2 + tid * DD;
        for (int k = 0; k < DD; ++k) acc += w[k] * hs[k];
        out[b * DD + tid] = acc;
    }

    // Per-sample A fragments (f16, fragment-ready, zero-padded)
    for (int s = tid; s < MT * KSTEPS * 32; s += 256) {
        int f = s >> 5, lane = s & 31;
        int m_tile = f >> 2, kstep = f & 3;
        int i = m_tile * 16 + (lane & 15);
        int base = f * FRAG_HALVES + lane * 16;
        for (int e = 0; e < 16; ++e) {
            int k = kstep * 32 + klocal(lane, e);
            float w2  = (i < DD && k < DD) ? W2[i * DD + k] : 0.f;
            float dd1 = (k < DD) ? d1s[k] : 0.f;
            float dd2 = (k < DD) ? d2s[k] : 0.f;
            A1[base + e] = (_Float16)(w2 * dd1);
            A2[base + e] = (_Float16)(w2 * dd2);
        }
    }
    __syncthreads();

    const int wave = tid >> 5, lane = tid & 31;
    const int cn = lane & 15;          // N within tile (C layout)
    const int cm = (lane >> 4) << 3;   // M base within tile (C layout)

    // -------- Jacobian: J = (W2 * g') @ W1, slice 0 only --------
    if (slice == 0) {
        const _Float16* Bw1 = wsB + (size_t)(NT * KSTEPS) * FRAG_HALVES + lane * 16;
        float* jacb = out + (size_t)BB * DD + (size_t)b * DD * DD;
        for (int t = wave; t < MT * MT; t += 8) {
            int mt = t / MT, nt = t % MT;      // wave-uniform
            v8f c = {};
            #pragma unroll
            for (int ks = 0; ks < KSTEPS; ++ks) {
                v16h a  = *(const v16h*)(A1 + (mt * KSTEPS + ks) * FRAG_HALVES + lane * 16);
                v16h bm = *(const v16h*)(Bw1 + (size_t)(nt * KSTEPS + ks) * FRAG_HALVES);
                c = wmma_f16(a, bm, c);
            }
            int j = nt * 16 + cn;
            float* p = jacb + (size_t)(mt * 16 + cm) * DD + j;
            bool jv = (nt < 6) || (cn < 4);    // j < 100
            if (mt < 6) {                      // wave-uniform branch
                if (jv) {
                    #pragma unroll
                    for (int r = 0; r < 8; ++r)
                        __builtin_nontemporal_store(c[r], p + (size_t)r * DD);
                }
            } else if (cm == 0 && jv) {        // i in 96..99 only
                #pragma unroll
                for (int r = 0; r < 4; ++r)
                    __builtin_nontemporal_store(c[r], p + (size_t)r * DD);
            }
        }
    }

    // -------- Hessian: H = (W2 * g'') @ P, this block's N slice --------
    float* hessb = out + (size_t)BB * DD + (size_t)BB * DD * DD
                       + (size_t)b * DD * DD * DD;
    const _Float16* Bl = wsB + lane * 16;

    for (int mt = 0; mt < MT; ++mt) {
        // A fragments for this mt stay in registers across all N-tiles
        v16h a0 = *(const v16h*)(A2 + (mt * KSTEPS + 0) * FRAG_HALVES + lane * 16);
        v16h a1 = *(const v16h*)(A2 + (mt * KSTEPS + 1) * FRAG_HALVES + lane * 16);
        v16h a2 = *(const v16h*)(A2 + (mt * KSTEPS + 2) * FRAG_HALVES + lane * 16);
        v16h a3 = *(const v16h*)(A2 + (mt * KSTEPS + 3) * FRAG_HALVES + lane * 16);
        float* pm = hessb + (size_t)(mt * 16 + cm) * DD * DD;

        int t = wave;
        // prologue: load first B-fragment set
        v16h b0, b1v, b2v, b3;
        if (t < NT_SLICE) {
            const _Float16* bp = Bl + (size_t)(slice * NT_SLICE + t) * KSTEPS * FRAG_HALVES;
            b0  = *(const v16h*)(bp);
            b1v = *(const v16h*)(bp + FRAG_HALVES);
            b2v = *(const v16h*)(bp + 2 * FRAG_HALVES);
            b3  = *(const v16h*)(bp + 3 * FRAG_HALVES);
        }
        while (t < NT_SLICE) {                 // wave-uniform trip count
            v16h c0 = b0, c1 = b1v, c2 = b2v, c3 = b3;
            int nt = slice * NT_SLICE + t;
            int t2 = t + 8;
            if (t2 < NT_SLICE) {               // prefetch next tile's B before WMMA
                const _Float16* bp = Bl + (size_t)(nt + 8) * KSTEPS * FRAG_HALVES;
                b0  = *(const v16h*)(bp);
                b1v = *(const v16h*)(bp + FRAG_HALVES);
                b2v = *(const v16h*)(bp + 2 * FRAG_HALVES);
                b3  = *(const v16h*)(bp + 3 * FRAG_HALVES);
            }
            v8f c = {};
            c = wmma_f16(a0, c0, c);
            c = wmma_f16(a1, c1, c);
            c = wmma_f16(a2, c2, c);
            c = wmma_f16(a3, c3, c);

            float* p = pm + nt * 16 + cn;      // n = j*100 + l, always < 10000
            if (mt < 6) {                      // wave-uniform: all 8 rows valid
                #pragma unroll
                for (int r = 0; r < 8; ++r)
                    __builtin_nontemporal_store(c[r], p + (size_t)r * DD * DD);
            } else if (cm == 0) {              // mt==6: only i = 96..99 valid
                #pragma unroll
                for (int r = 0; r < 4; ++r)
                    __builtin_nontemporal_store(c[r], p + (size_t)r * DD * DD);
            }
            t = t2;
        }
    }
}

extern "C" void kernel_launch(void* const* d_in, const int* in_sizes, int n_in,
                              void* d_out, int out_size, void* d_ws, size_t ws_size,
                              hipStream_t stream) {
    const float* x  = (const float*)d_in[0];
    const float* W1 = (const float*)d_in[1];
    const float* b1 = (const float*)d_in[2];
    const float* W2 = (const float*)d_in[3];
    const float* b2 = (const float*)d_in[4];
    _Float16* ws = (_Float16*)d_ws;   // needs 2,588,672 bytes

    const int prepSlots = NT * KSTEPS * 32 + MT * KSTEPS * 32;  // 80896
    diffn_prep<<<(prepSlots + 255) / 256, 256, 0, stream>>>(W1, ws);

    dim3 grid(BB, NSLICE);
    diffn_ffn<<<grid, 256, 0, stream>>>(x, W1, b1, W2, b2, ws, (float*)d_out);
}
